// SymmetricRBM_73830487818492
// MI455X (gfx1250) — matrix operand order repaired and checked by reading, not compiled, not run on hardware
//
#include <hip/hip_runtime.h>
#include <hip/hip_bf16.h>
#include <stdint.h>

// ---------------------------------------------------------------------------
// CDNA5 (gfx1250) wave32 WMMA types
// ---------------------------------------------------------------------------
typedef __attribute__((ext_vector_type(16))) _Float16 v16h;
typedef __attribute__((ext_vector_type(8)))  _Float16 v8h;
typedef __attribute__((ext_vector_type(8)))  float    v8f;

#define TILE_M 128
#define TILE_N 128
#define TILE_K 32
#define LDA_SH 40   // padded half-stride (80B rows): conflict-free b128 LDS reads
#define LDB_SH 40   // same padding baked into the pre-swizzled global W image

// ---------------------------------------------------------------------------
// Stateless counter RNG (deterministic across graph replays)
// ---------------------------------------------------------------------------
__device__ __forceinline__ uint32_t hash4(uint32_t a, uint32_t b, uint32_t c, uint32_t d) {
  uint32_t x = a * 0x9E3779B9u;
  x ^= (b + 0x7F4A7C15u) * 0x85EBCA6Bu;
  x ^= (c + 0x165667B1u) * 0xC2B2AE35u;
  x ^= (d + 0x27D4EB2Fu) * 0x9E3779B1u;
  x ^= x >> 16; x *= 0x7FEB352Du;
  x ^= x >> 15; x *= 0x846CA68Bu;
  x ^= x >> 16;
  return x;
}
__device__ __forceinline__ float u01(uint32_t h) {
  return (float)(h >> 8) * (1.0f / 16777216.0f);
}
__device__ __forceinline__ float sigm(float x) { return 1.0f / (1.0f + __expf(-x)); }
__device__ __forceinline__ float softplus_f(float x) {
  return fmaxf(x, 0.0f) + log1pf(__expf(-fabsf(x)));
}

// ---------------------------------------------------------------------------
// WMMA GEMM: C[M,N] = A[M,K](f16, row-major) x B[K,N](f16, pre-swizzled tiles)
// Block tile 128x128, 8 waves (2M x 4N), wave tile 64x32 -> 8 WMMAs / K-step.
// Tiles double-buffered in LDS via GLOBAL_LOAD_ASYNC_TO_LDS_B128 (ASYNCcnt).
// EPI 0: Out = bernoulli(sigmoid(C + c[col]))            (h sampling)
// EPI 1: Out = C; rsum1 += sum_n C, rsum2 += sum_n C*Vc  (a + row sums)
// EPI 2: rsum1 += sum softplus(C+c), rsum2 += sum softplus(wcs-C+c)
// ---------------------------------------------------------------------------
template <int EPI>
__global__ __launch_bounds__(256) void gemm_wmma(
    const _Float16* __restrict__ A,
    const _Float16* __restrict__ Bsw,   // swizzled: [nt][kt][128][LDB_SH]
    const float*    __restrict__ cvec,
    const float*    __restrict__ wcs,
    _Float16*       __restrict__ Out,
    const _Float16* __restrict__ Vc,
    float*          __restrict__ rsum1,
    float*          __restrict__ rsum2,
    const int*      __restrict__ seedp,
    int step, int Ndim, int Kdim) {
  __shared__ __align__(16) _Float16 sA[2][TILE_M * LDA_SH];
  __shared__ __align__(16) _Float16 sB[2][TILE_N * LDB_SH];

  const int tid   = threadIdx.x;
  const int lane  = tid & 31;
  const int wave  = tid >> 5;
  const int wm    = wave & 1;    // 0..1 : 64-row slab
  const int wn    = wave >> 1;   // 0..3 : 32-col slab
  const int mlane = lane & 15;
  const int khalf = lane >> 4;
  const int rowBase = blockIdx.x * TILE_M;
  const int colBase = blockIdx.y * TILE_N;
  const int KT = Kdim / TILE_K;
  const uint32_t seed = (uint32_t)seedp[0];

  const uint32_t sA0 = (uint32_t)(uintptr_t)&sA[0][0];
  const uint32_t sB0 = (uint32_t)(uintptr_t)&sB[0][0];
  const size_t tileBytesB = (size_t)TILE_N * LDB_SH * 2;  // 10240 B

  // issue one (A,B) tile pair as async global->LDS copies (5 b128 per thread)
  auto issue_tile = [&](int buf, int kt) {
    // ---- A tile: 128 rows x 64B data per row, padded 80B rows: 512 chunks ----
    const uint32_t ldsA = sA0 + (uint32_t)buf * (TILE_M * LDA_SH * 2);
    const char* Ab = (const char*)A;
#pragma unroll
    for (int j = 0; j < 2; ++j) {
      int ch = tid + j * 256;
      int row = ch >> 2, part = ch & 3;
      uint32_t ldst = ldsA + (uint32_t)(row * (LDA_SH * 2) + part * 16);
      uint64_t gsrc = (uint64_t)(uintptr_t)(Ab +
          ((size_t)(rowBase + row) * Kdim + (size_t)kt * TILE_K + part * 8) * 2);
      asm volatile("global_load_async_to_lds_b128 %0, %1, off"
                   :: "v"(ldst), "v"(gsrc) : "memory");
    }
    // ---- B tile: verbatim 10240B contiguous copy: 640 chunks ----
    const uint32_t ldsB = sB0 + (uint32_t)buf * (TILE_N * LDB_SH * 2);
    const char* Bb = (const char*)Bsw +
        ((size_t)blockIdx.y * KT + kt) * tileBytesB;
#pragma unroll
    for (int j = 0; j < 2; ++j) {
      int ch = tid + j * 256;
      asm volatile("global_load_async_to_lds_b128 %0, %1, off"
                   :: "v"(ldsB + (uint32_t)(ch * 16)),
                      "v"((uint64_t)(uintptr_t)(Bb + (size_t)ch * 16)) : "memory");
    }
    if ((tid & 1) == 0) {  // 128 extra chunks spread across all waves
      int ch = 512 + (tid >> 1);
      asm volatile("global_load_async_to_lds_b128 %0, %1, off"
                   :: "v"(ldsB + (uint32_t)(ch * 16)),
                      "v"((uint64_t)(uintptr_t)(Bb + (size_t)ch * 16)) : "memory");
    }
  };

  v8f acc[4][2] = {};

  issue_tile(0, 0);  // prologue

  for (int it = 0; it < KT; ++it) {
    __syncthreads();  // everyone done reading the buffer we are about to refill
    if (it + 1 < KT) {
      issue_tile((it + 1) & 1, it + 1);
      // in-order completion: <=5 outstanding => current tile fully in LDS
      asm volatile("s_wait_asynccnt 0x5" ::: "memory");
    } else {
      asm volatile("s_wait_asynccnt 0x0" ::: "memory");
    }
    __syncthreads();  // all waves' async data visible

    const _Float16* sAb = &sA[it & 1][0];
    const _Float16* sBb = &sB[it & 1][0];

    v16h afrag[4], bfrag[2];
#pragma unroll
    for (int mi = 0; mi < 4; ++mi) {
      // A 16x32 f16 frag: lanes 0-15 K[0..7]/[16..23]; lanes 16-31 K[8..15]/[24..31]
      const _Float16* ap = sAb + (wm * 64 + mi * 16 + mlane) * LDA_SH + khalf * 8;
      v8h lo = *(const v8h*)(ap);
      v8h hi = *(const v8h*)(ap + 16);
      afrag[mi] = __builtin_shufflevector(lo, hi, 0,1,2,3,4,5,6,7,8,9,10,11,12,13,14,15);
    }
#pragma unroll
    for (int ni = 0; ni < 2; ++ni) {
      // B 32x16 f16 frag: lanes 0-15 K[0..7]/[16..23]; lanes 16-31 K[8..15]/[24..31]
      const _Float16* bp = sBb + (wn * 32 + ni * 16 + mlane) * LDB_SH + khalf * 8;
      v8h lo = *(const v8h*)(bp);
      v8h hi = *(const v8h*)(bp + 16);
      bfrag[ni] = __builtin_shufflevector(lo, hi, 0,1,2,3,4,5,6,7,8,9,10,11,12,13,14,15);
    }
#pragma unroll
    for (int mi = 0; mi < 4; ++mi)
#pragma unroll
      for (int ni = 0; ni < 2; ++ni)
        acc[mi][ni] = __builtin_amdgcn_wmma_f32_16x16x32_f16(
            false, afrag[mi], false, bfrag[ni], (short)0, acc[mi][ni], false, false);
  }

  // ---- epilogues (C/D layout: VGPR rr -> row khalf*8+rr, col mlane) ----
  if (EPI == 0) {
#pragma unroll
    for (int mi = 0; mi < 4; ++mi)
#pragma unroll
      for (int ni = 0; ni < 2; ++ni)
#pragma unroll
        for (int rr = 0; rr < 8; ++rr) {
          int row = rowBase + wm * 64 + mi * 16 + khalf * 8 + rr;
          int col = colBase + wn * 32 + ni * 16 + mlane;
          float p = sigm(acc[mi][ni][rr] + cvec[col]);
          float r = u01(hash4(seed, 0x48000000u + (uint32_t)step, (uint32_t)row, (uint32_t)col));
          Out[(size_t)row * Ndim + col] = (_Float16)((r < p) ? 1.0f : 0.0f);
        }
  } else if (EPI == 1) {
#pragma unroll
    for (int mi = 0; mi < 4; ++mi)
#pragma unroll
      for (int rr = 0; rr < 8; ++rr) {
        int row = rowBase + wm * 64 + mi * 16 + khalf * 8 + rr;
        float s1 = 0.0f, s2 = 0.0f;
#pragma unroll
        for (int ni = 0; ni < 2; ++ni) {
          int col = colBase + wn * 32 + ni * 16 + mlane;
          float val = acc[mi][ni][rr];
          Out[(size_t)row * Ndim + col] = (_Float16)val;
          s1 += val;
          s2 += val * (float)Vc[(size_t)row * Ndim + col];
        }
#pragma unroll
        for (int off = 1; off < 16; off <<= 1) {  // reduce across the 16 n-lanes
          s1 += __shfl_xor(s1, off, 32);
          s2 += __shfl_xor(s2, off, 32);
        }
        if (mlane == 0) {
          atomicAdd(&rsum1[row], s1);
          atomicAdd(&rsum2[row], s2);
        }
      }
  } else {  // EPI == 2 : free-energy softplus partial sums
#pragma unroll
    for (int mi = 0; mi < 4; ++mi)
#pragma unroll
      for (int rr = 0; rr < 8; ++rr) {
        int row = rowBase + wm * 64 + mi * 16 + khalf * 8 + rr;
        float s1 = 0.0f, s2 = 0.0f;
#pragma unroll
        for (int ni = 0; ni < 2; ++ni) {
          int col = colBase + wn * 32 + ni * 16 + mlane;
          float val = acc[mi][ni][rr];
          s1 += softplus_f(val + cvec[col]);
          s2 += softplus_f(wcs[col] - val + cvec[col]);
        }
#pragma unroll
        for (int off = 1; off < 16; off <<= 1) {
          s1 += __shfl_xor(s1, off, 32);
          s2 += __shfl_xor(s2, off, 32);
        }
        if (mlane == 0) {
          atomicAdd(&rsum1[row], s1);
          atomicAdd(&rsum2[row], s2);
        }
      }
  }
}

// ---------------------------------------------------------------------------
// Pre-swizzle W (or W^T) into B-fragment-ready padded tiles:
//   out[((nt*KT + kt)*128 + nl)*LDB_SH + kl] = B[kt*32+kl][nt*128+nl], kl<32
// ---------------------------------------------------------------------------
__global__ void k_swz(const float* __restrict__ W, _Float16* __restrict__ out,
                      int total, int KT, int transpose) {
  int flat = blockIdx.x * blockDim.x + threadIdx.x;
  if (flat < total) {
    int kl = flat % LDB_SH;
    int r  = flat / LDB_SH;
    int nl = r % TILE_N;
    int r2 = r / TILE_N;
    int kt = r2 % KT;
    int nt = r2 / KT;
    int kk = kt * TILE_K + kl;
    int nn = nt * TILE_N + nl;
    _Float16 v = (_Float16)0;
    if (kl < TILE_K)
      v = (_Float16)(transpose ? W[(size_t)nn * 1024 + kk]
                               : W[(size_t)kk * 1024 + nn]);
    out[flat] = v;
  }
}

__global__ void k_colsum(const float* __restrict__ W, float* __restrict__ wcs, int V, int H) {
  int j = blockIdx.x * blockDim.x + threadIdx.x;
  if (j < H) {
    float s = 0.0f;
    for (int i = 0; i < V; ++i) s += W[(size_t)i * H + j];
    wcs[j] = s;
  }
}

__global__ void k_bsum(const float* __restrict__ b, float* __restrict__ out, int n) {
  __shared__ float sm[256];
  float s = 0.0f;
  for (int i = threadIdx.x; i < n; i += 256) s += b[i];
  sm[threadIdx.x] = s;
  __syncthreads();
  for (int w = 128; w > 0; w >>= 1) {
    if (threadIdx.x < (unsigned)w) sm[threadIdx.x] += sm[threadIdx.x + w];
    __syncthreads();
  }
  if (threadIdx.x == 0) out[0] = sm[0];
}

__global__ void k_init_u(float* __restrict__ u, int n, const int* __restrict__ seedp) {
  int i = blockIdx.x * blockDim.x + threadIdx.x;
  if (i < n)
    u[i] = (u01(hash4((uint32_t)seedp[0], 0x55AA55AAu, (uint32_t)i, 0u)) < 0.5f) ? 1.0f : 0.0f;
}

// v_data -> f16 copies + v_branch0 = u0*v + (1-u0)*(1-v)
__global__ void k_init_v(const float* __restrict__ vd, const float* __restrict__ u,
                         _Float16* __restrict__ vdh, _Float16* __restrict__ vch,
                         _Float16* __restrict__ vbr, int n, int V) {
  int i = blockIdx.x * blockDim.x + threadIdx.x;
  if (i < n) {
    float x = vd[i];
    _Float16 h = (_Float16)x;
    vdh[i] = h;
    vch[i] = h;
    vbr[i] = (_Float16)((u[i / V] > 0.5f) ? x : 1.0f - x);
  }
}

// One wave per batch row: dE, u_new, v_new, combine; also writes next v_branch.
__global__ __launch_bounds__(256) void k_update(
    _Float16* __restrict__ vcur, _Float16* __restrict__ vbr,
    const _Float16* __restrict__ a_h,
    const float* __restrict__ bvec, const float* __restrict__ arow,
    const float* __restrict__ varow, const float* __restrict__ bsum,
    float* __restrict__ u, const int* __restrict__ seedp, int step, int N) {
  int lane = threadIdx.x & 31;
  int row  = blockIdx.x * 8 + (threadIdx.x >> 5);
  uint32_t seed = (uint32_t)seedp[0];
  float vb = 0.0f;
  for (int k2 = lane; k2 < N; k2 += 32)
    vb += (float)vcur[(size_t)row * N + k2] * bvec[k2];
#pragma unroll
  for (int off = 16; off >= 1; off >>= 1) vb += __shfl_xor(vb, off, 32);
  float dE = -bsum[0] - arow[row] + 2.0f * vb + 2.0f * varow[row];
  float un = (u01(hash4(seed, 0x21000000u + (uint32_t)step, (uint32_t)row, 0xFFFFu)) < sigm(dE))
                 ? 1.0f : 0.0f;
  for (int k2 = lane; k2 < N; k2 += 32) {
    float av = (float)a_h[(size_t)row * N + k2];
    float p  = sigm(av + bvec[k2]);
    float vn = (u01(hash4(seed, 0x37000000u + (uint32_t)step, (uint32_t)row, (uint32_t)k2)) < p)
                   ? 1.0f : 0.0f;
    float vfin = (un > 0.5f) ? vn : 1.0f - vn;
    vcur[(size_t)row * N + k2] = (_Float16)vfin;
    vbr[(size_t)row * N + k2] = (_Float16)((un > 0.5f) ? vfin : 1.0f - vfin);
  }
  if (lane == 0) u[row] = un;
}

// One wave per row: F = -logsumexp([vb - sp1, (bsum - vb) - sp2]); out += sign*F/B
__global__ __launch_bounds__(256) void k_fe_final(
    const _Float16* __restrict__ v, const float* __restrict__ bvec,
    const float* __restrict__ sp1, const float* __restrict__ sp2,
    const float* __restrict__ bsum, float* __restrict__ out,
    float sign, int N, int Brows) {
  int lane = threadIdx.x & 31;
  int row  = blockIdx.x * 8 + (threadIdx.x >> 5);
  float vb = 0.0f;
  for (int k2 = lane; k2 < N; k2 += 32)
    vb += (float)v[(size_t)row * N + k2] * bvec[k2];
#pragma unroll
  for (int off = 16; off >= 1; off >>= 1) vb += __shfl_xor(vb, off, 32);
  if (lane == 0) {
    float nfn = vb - sp1[row];
    float nff = (bsum[0] - vb) - sp2[row];
    float m = fmaxf(nfn, nff);
    float F = -(m + logf(__expf(nfn - m) + __expf(nff - m)));
    atomicAdd(out, sign * F / (float)Brows);
  }
}

// ---------------------------------------------------------------------------
// Launch
// ---------------------------------------------------------------------------
extern "C" void kernel_launch(void* const* d_in, const int* in_sizes, int n_in,
                              void* d_out, int out_size, void* d_ws, size_t ws_size,
                              hipStream_t stream) {
  const float* v_data = (const float*)d_in[0];
  const float* W      = (const float*)d_in[1];
  const float* b      = (const float*)d_in[2];
  const float* c      = (const float*)d_in[3];
  const int*   seedp  = (const int*)d_in[4];
  (void)in_sizes; (void)n_in; (void)ws_size;

  const int Brows = 8192, V = 1024, H = 1024, KSTEPS = 10;
  const int KT = V / TILE_K;                       // 32
  const int swzTotal = (H / TILE_N) * KT * TILE_N * LDB_SH;  // 1,310,720 halves

  char* ws = (char*)d_ws;
  size_t off = 0;
  auto alloc = [&](size_t bytes) -> void* {
    void* p = (void*)(ws + off);
    off += (bytes + 255) & ~(size_t)255;
    return p;
  };
  _Float16* vdh  = (_Float16*)alloc((size_t)Brows * V * 2);  // v_data (f16)
  _Float16* vch  = (_Float16*)alloc((size_t)Brows * V * 2);  // current v
  _Float16* vbr  = (_Float16*)alloc((size_t)Brows * V * 2);  // v_branch
  _Float16* h_h  = (_Float16*)alloc((size_t)Brows * H * 2);  // hidden samples
  _Float16* a_h  = (_Float16*)alloc((size_t)Brows * V * 2);  // a = h @ W^T
  _Float16* swzW  = (_Float16*)alloc((size_t)swzTotal * 2);  // W   as B tiles
  _Float16* swzWT = (_Float16*)alloc((size_t)swzTotal * 2);  // W^T as B tiles
  float* u    = (float*)alloc((size_t)Brows * 4);
  float* arow = (float*)alloc((size_t)Brows * 4);  // contiguous with varow
  float* varw = (float*)alloc((size_t)Brows * 4);
  float* sp1  = (float*)alloc((size_t)Brows * 4);  // contiguous with sp2
  float* sp2  = (float*)alloc((size_t)Brows * 4);
  float* wcs  = (float*)alloc((size_t)H * 4);
  float* bsum = (float*)alloc(256);

  hipMemsetAsync(d_out, 0, (size_t)out_size * sizeof(float), stream);

  // prep
  k_swz<<<(swzTotal + 255) / 256, 256, 0, stream>>>(W, swzW,  swzTotal, KT, 0);
  k_swz<<<(swzTotal + 255) / 256, 256, 0, stream>>>(W, swzWT, swzTotal, KT, 1);
  k_colsum<<<(H + 255) / 256, 256, 0, stream>>>(W, wcs, V, H);
  k_bsum<<<1, 256, 0, stream>>>(b, bsum, V);
  k_init_u<<<(Brows + 255) / 256, 256, 0, stream>>>(u, Brows, seedp);
  k_init_v<<<(Brows * V + 255) / 256, 256, 0, stream>>>(v_data, u, vdh, vch, vbr,
                                                        Brows * V, V);

  dim3 gg(Brows / TILE_M, H / TILE_N);  // 64 x 8
  dim3 gb(256);

  // Gibbs chain
  for (int t = 0; t < KSTEPS; ++t) {
    // h = bern(sigmoid(v_branch @ W + c))
    gemm_wmma<0><<<gg, gb, 0, stream>>>(vbr, swzW, c, wcs, h_h, vch,
                                        nullptr, nullptr, seedp, t, H, V);
    hipMemsetAsync(arow, 0, 2 * (size_t)Brows * sizeof(float), stream);
    // a = h @ W^T, fused row sums of a and v*a
    gemm_wmma<1><<<gg, gb, 0, stream>>>(h_h, swzWT, c, wcs, a_h, vch,
                                        arow, varw, seedp, t, V, H);
    // dE, u_new, v_new, combine; writes vcur and next v_branch
    k_update<<<Brows / 8, 256, 0, stream>>>(vch, vbr, a_h, b, arow, varw, bsum,
                                            u, seedp, t, V);
  }

  // free energy of v_data (+) and v_model (-)
  hipMemsetAsync(sp1, 0, 2 * (size_t)Brows * sizeof(float), stream);
  gemm_wmma<2><<<gg, gb, 0, stream>>>(vdh, swzW, c, wcs, nullptr, vch,
                                      sp1, sp2, seedp, 0, H, V);
  k_fe_final<<<Brows / 8, 256, 0, stream>>>(vdh, b, sp1, sp2, bsum,
                                            (float*)d_out, 1.0f, V, Brows);

  hipMemsetAsync(sp1, 0, 2 * (size_t)Brows * sizeof(float), stream);
  gemm_wmma<2><<<gg, gb, 0, stream>>>(vch, swzW, c, wcs, nullptr, vch,
                                      sp1, sp2, seedp, 1, H, V);
  k_fe_final<<<Brows / 8, 256, 0, stream>>>(vch, b, sp1, sp2, bsum,
                                            (float*)d_out, -1.0f, V, Brows);
}
